// MultiGATNet_56057913147918
// MI455X (gfx1250) — matrix operand order repaired and checked by reading, not compiled, not run on hardware
//
#include <hip/hip_runtime.h>

// ---------------------------------------------------------------------------
// MultiGATNet for MI455X (gfx1250): bf16 WMMA GEMMs + atomic scatter attention
// ---------------------------------------------------------------------------

typedef __attribute__((ext_vector_type(16))) __bf16 v16bf;
typedef __attribute__((ext_vector_type(8)))  float  v8f;

#define BMT 128
#define BNT 64
#define BKT 32
#define LDSTR 40   // padded LDS row stride (bf16 elems): 16B-aligned rows, conflict-free

// fragment gather + 4x WMMA for one 128x64x32 block step (one wave = 16x64 strip)
__device__ __forceinline__ void gemm_mma_step(const __bf16* As, const __bf16* Bs,
                                              int wave, int lane, v8f acc[4]) {
  // A (16x32): row = lane&15, half = 8*(lane>=16);
  //   elem e<8 -> K=half+e ; e>=8 -> K=16+half+(e-8)  (two 16B chunks)
  int r    = wave * 16 + (lane & 15);
  int half = (lane >> 4) * 8;
  const __bf16* ap = &As[r * LDSTR + half];
  v16bf a;
#pragma unroll
  for (int e = 0; e < 8; ++e) { a[e] = ap[e]; a[8 + e] = ap[16 + e]; }

  // B (32x16): col = lane&15, khalf = 8*(lane>=16), elem e<8 -> K=khalf+e ; e>=8 -> K=16+khalf+(e-8)
  int khalf = (lane >> 4) * 8;
  int nlane = lane & 15;
  v16bf b[4];
#pragma unroll
  for (int t = 0; t < 4; ++t) {
    const __bf16* bp = &Bs[(t * 16 + nlane) * LDSTR + khalf];
#pragma unroll
    for (int e = 0; e < 8; ++e) { b[t][e] = bp[e]; b[t][8 + e] = bp[16 + e]; }
  }
#pragma unroll
  for (int t = 0; t < 4; ++t)
    acc[t] = __builtin_amdgcn_wmma_f32_16x16x32_bf16(false, a, false, b[t],
                                                     (short)0, acc[t], false, false);
}

// ---------------- GEMM: C[m, colOff+n] = act(A[M,K] @ W[K,N] + bias[n]) -----
// act: 0 = none, 1 = relu, 2 = elu
__global__ __launch_bounds__(256) void k_gemm(const float* __restrict__ A,
                                              const float* __restrict__ W,
                                              const float* __restrict__ bias,
                                              float* __restrict__ C,
                                              int M, int N, int K,
                                              int ldc, int colOff, int act) {
  __shared__ __bf16 As[BMT * LDSTR];   // [m][k], row-major
  __shared__ __bf16 Bs[BNT * LDSTR];   // [n][k], transposed from W
  const int tid  = threadIdx.x;
  const int lane = tid & 31;
  const int wave = tid >> 5;           // 8 waves -> 8 x 16 rows = 128 M
  const int m0 = blockIdx.y * BMT;
  const int n0 = blockIdx.x * BNT;

  v8f acc[4] = {{}, {}, {}, {}};

  const bool edge_mn = (m0 + BMT > M) || (n0 + BNT > N);
  const int  kMain   = edge_mn ? 0 : (K & ~(BKT - 1));

  // per-thread staging coordinates (constant across K steps)
  const int aM[8] = { (tid + 0*256) >> 4, (tid + 1*256) >> 4, (tid + 2*256) >> 4,
                      (tid + 3*256) >> 4, (tid + 4*256) >> 4, (tid + 5*256) >> 4,
                      (tid + 6*256) >> 4, (tid + 7*256) >> 4 };
  const int aKP = (tid & 15) << 1;                 // same for all i
  const int bN  = tid & 63, bKQ = (tid >> 6) << 3; // B: column + k-quad base

  int k0 = 0;
  // =================== fast main loop: no guards ===================
  for (; k0 < kMain; k0 += BKT) {
    // ---- batched global loads (all issued before any use) ----
    float a0[8], a1[8];
#pragma unroll
    for (int i = 0; i < 8; ++i) {
      const float* ap = &A[(size_t)(m0 + aM[i]) * K + (k0 + aKP)];
      a0[i] = ap[0]; a1[i] = ap[1];
    }
    float bv[8];
    {
      const float* wp = &W[(size_t)(k0 + bKQ) * N + (n0 + bN)];
#pragma unroll
      for (int j = 0; j < 8; ++j) bv[j] = wp[(size_t)j * N];
    }
    // ---- convert + LDS stores ----
#pragma unroll
    for (int i = 0; i < 8; ++i) {
      union { __bf16 h[2]; unsigned u; } p;
      p.h[0] = (__bf16)a0[i]; p.h[1] = (__bf16)a1[i];
      *(unsigned*)&As[aM[i] * LDSTR + aKP] = p.u;
    }
    {
      union { __bf16 h[8]; uint4 q; } p;
#pragma unroll
      for (int j = 0; j < 8; ++j) p.h[j] = (__bf16)bv[j];
      *(uint4*)&Bs[bN * LDSTR + bKQ] = p.q;
    }
    // prefetch next K-tile of the streamed weight matrix
    if (k0 + BKT < K)
      __builtin_prefetch(&W[(size_t)(k0 + BKT + bKQ) * N + (n0 + bN)], 0, 0);

    __syncthreads();
    gemm_mma_step(As, Bs, wave, lane, acc);
    __syncthreads();
  }

  // =================== guarded tail loop (<= 1 iter unless edge block) ======
  for (; k0 < K; k0 += BKT) {
#pragma unroll
    for (int i = 0; i < 16; ++i) {      // A: 128*32 elems
      int idx = tid + i * 256;
      int m = idx >> 5, k = idx & 31;
      int gm = m0 + m, gk = k0 + k;
      float v = (gm < M && gk < K) ? A[(size_t)gm * K + gk] : 0.f;
      As[m * LDSTR + k] = (__bf16)v;
    }
#pragma unroll
    for (int i = 0; i < 8; ++i) {       // B: 64*32 elems
      int idx = tid + i * 256;
      int k = idx >> 6, n = idx & 63;
      int gk = k0 + k, gn = n0 + n;
      float v = (gk < K && gn < N) ? W[(size_t)gk * N + gn] : 0.f;
      Bs[n * LDSTR + k] = (__bf16)v;
    }
    __syncthreads();
    gemm_mma_step(As, Bs, wave, lane, acc);
    __syncthreads();
  }

  // --- epilogue: C/D layout: VGPR r8 -> M = r8 + 8*(lane>=16), N = lane&15 ---
  int mrBase = wave * 16 + 8 * (lane >> 4);
  int nlane  = lane & 15;
#pragma unroll
  for (int t = 0; t < 4; ++t) {
    int gn = n0 + t * 16 + nlane;
    if (gn >= N) continue;
    float bvv = bias ? bias[gn] : 0.f;
#pragma unroll
    for (int r8 = 0; r8 < 8; ++r8) {
      int gm = m0 + mrBase + r8;
      if (gm >= M) continue;
      float v = acc[t][r8] + bvv;
      if (act == 1)      v = v > 0.f ? v : 0.f;
      else if (act == 2) v = v > 0.f ? v : (__expf(v) - 1.f);
      C[(size_t)gm * ldc + colOff + gn] = v;
    }
  }
}

// ---------------- row L2 normalize: out = x / max(||x||_2, 1e-12) ----------
__global__ __launch_bounds__(256) void k_l2norm(const float* __restrict__ in,
                                                float* __restrict__ out, int cols) {
  __shared__ float red[256];
  int row = blockIdx.x;
  const float* ip = in + (size_t)row * cols;
  float* op = out + (size_t)row * cols;
  float s = 0.f;
  for (int c = threadIdx.x; c < cols; c += 256) { float v = ip[c]; s += v * v; }
  red[threadIdx.x] = s;
  __syncthreads();
  for (int st = 128; st > 0; st >>= 1) {
    if (threadIdx.x < st) red[threadIdx.x] += red[threadIdx.x + st];
    __syncthreads();
  }
  float inv = 1.f / fmaxf(sqrtf(red[0]), 1e-12f);
  for (int c = threadIdx.x; c < cols; c += 256) op[c] = ip[c] * inv;
}

// ---------------- misc elementwise ----------------
__global__ void k_fill(float* p, float v, int n) {
  int i = blockIdx.x * blockDim.x + threadIdx.x;
  if (i < n) p[i] = v;
}
__global__ void k_bias_rows(float* __restrict__ out, const float* __restrict__ b,
                            int rows, int cols) {
  int i = blockIdx.x * blockDim.x + threadIdx.x;
  if (i < rows * cols) out[i] = b[i % cols];
}
__global__ void k_elu(float* x, int n) {
  int i = blockIdx.x * blockDim.x + threadIdx.x;
  if (i < n) { float v = x[i]; x[i] = v > 0.f ? v : (__expf(v) - 1.f); }
}

// ---------------- GAT attention ----------------
// asrc[n,h] = sum_d H[n,h,d]*a_src[h,d] ; adst analog
__global__ void k_att_scores(const float* __restrict__ H,
                             const float* __restrict__ a_src,
                             const float* __restrict__ a_dst,
                             float* __restrict__ asrc, float* __restrict__ adst,
                             int N, int Hh, int D) {
  int i = blockIdx.x * blockDim.x + threadIdx.x;
  if (i >= N * Hh) return;
  int n = i / Hh, h = i % Hh;
  const float* hp = H + (size_t)n * Hh * D + (size_t)h * D;
  const float* s = a_src + (size_t)h * D;
  const float* d = a_dst + (size_t)h * D;
  float s1 = 0.f, s2 = 0.f;
  for (int k = 0; k < D; ++k) { float v = hp[k]; s1 += v * s[k]; s2 += v * d[k]; }
  asrc[i] = s1; adst[i] = s2;
}

__device__ __forceinline__ void atomicMaxF(float* a, float v) {
  unsigned int old = __float_as_uint(*a);
  while (v > __uint_as_float(old)) {
    unsigned int assumed = old;
    old = atomicCAS((unsigned int*)a, assumed, __float_as_uint(v));
    if (old == assumed) break;
  }
}
__device__ __forceinline__ void edge_src_dst(const int* ei, int E, int e,
                                             int& src, int& dst) {
  if (e < E) { src = ei[e]; dst = ei[E + e]; }
  else       { src = e - E; dst = e - E; }   // self loop
}
__device__ __forceinline__ float lrelu02(float x) { return x > 0.f ? x : 0.2f * x; }

// pass 1: segment max of leaky_relu logits per (dst, head)
__global__ void k_edge_max(const int* __restrict__ ei, int E, int ET, int Hh,
                           const float* __restrict__ asrc,
                           const float* __restrict__ adst,
                           float* __restrict__ mbuf) {
  int i = blockIdx.x * blockDim.x + threadIdx.x;
  if (i >= ET * Hh) return;
  int e = i / Hh, h = i % Hh, src, dst;
  edge_src_dst(ei, E, e, src, dst);
  float l = lrelu02(asrc[src * Hh + h] + adst[dst * Hh + h]);
  atomicMaxF(&mbuf[dst * Hh + h], l);
}
// pass 2: segment sum of exp(logit - m)
__global__ void k_edge_sum(const int* __restrict__ ei, int E, int ET, int Hh,
                           const float* __restrict__ asrc,
                           const float* __restrict__ adst,
                           const float* __restrict__ mbuf,
                           float* __restrict__ sbuf) {
  int i = blockIdx.x * blockDim.x + threadIdx.x;
  if (i >= ET * Hh) return;
  int e = i / Hh, h = i % Hh, src, dst;
  edge_src_dst(ei, E, e, src, dst);
  float l = lrelu02(asrc[src * Hh + h] + adst[dst * Hh + h]);
  atomicAdd(&sbuf[dst * Hh + h], __expf(l - mbuf[dst * Hh + h]));
}
// pass 3: out[dst,h,:] += H[src,h,:] * alpha  (out pre-filled with bias)
__global__ __launch_bounds__(128) void k_edge_apply(const int* __restrict__ ei,
    int E, int ET, int Hh, int D,
    const float* __restrict__ asrc, const float* __restrict__ adst,
    const float* __restrict__ mbuf, const float* __restrict__ sbuf,
    const float* __restrict__ H, float* __restrict__ out) {
  int eh = blockIdx.x;
  int e = eh / Hh, h = eh % Hh, src, dst;
  edge_src_dst(ei, E, e, src, dst);
  float l = lrelu02(asrc[src * Hh + h] + adst[dst * Hh + h]);
  float alpha = __expf(l - mbuf[dst * Hh + h]) / (sbuf[dst * Hh + h] + 1e-16f);
  int d = threadIdx.x;
  atomicAdd(&out[(size_t)dst * Hh * D + (size_t)h * D + d],
            H[(size_t)src * Hh * D + (size_t)h * D + d] * alpha);
}

// ---------------- global max pool by graph ----------------
__global__ void k_pool_max(const float* __restrict__ h, const int* __restrict__ batch,
                           float* __restrict__ pool, int N, int D) {
  int i = blockIdx.x * blockDim.x + threadIdx.x;
  if (i >= N * D) return;
  int n = i / D, d = i % D;
  atomicMaxF(&pool[(size_t)batch[n] * D + d], h[i]);
}
__global__ void k_pool_fix(float* pool, int n) {
  int i = blockIdx.x * blockDim.x + threadIdx.x;
  if (i < n) { float v = pool[i]; pool[i] = (v < -1e29f) ? 0.f : v; }
}

// ---------------- tiny output layer: [B,128] @ [128,2] + b ----------------
__global__ void k_out2(const float* __restrict__ X, const float* __restrict__ W,
                       const float* __restrict__ b, float* __restrict__ out,
                       int Bn, int K, int On) {
  int i = blockIdx.x * blockDim.x + threadIdx.x;
  if (i >= Bn * On) return;
  int r = i / On, j = i % On;
  float s = b[j];
  for (int k = 0; k < K; ++k) s += X[(size_t)r * K + k] * W[k * On + j];
  out[i] = s;
}

// ---------------------------------------------------------------------------
extern "C" void kernel_launch(void* const* d_in, const int* in_sizes, int n_in,
                              void* d_out, int out_size, void* d_ws, size_t ws_size,
                              hipStream_t stream) {
  (void)n_in; (void)ws_size;
  const int Bn  = out_size / 2;          // 512 graphs
  const int Nn  = in_sizes[2];           // 20480 nodes
  const int E   = in_sizes[1] / 2;       // 81920 edges
  const int ET  = E + Nn;                // + self loops
  const int F   = in_sizes[0] / Nn;      // 78
  const int D   = 128;
  const int HD1 = in_sizes[13];          // 1280
  const int Hh1 = HD1 / D;               // 10 heads

  const float* x_[2]   = {(const float*)d_in[0], (const float*)d_in[3]};
  const int*   ei_[2]  = {(const int*)d_in[1],  (const int*)d_in[4]};
  const int*   bat_[2] = {(const int*)d_in[2],  (const int*)d_in[5]};
  const float* cell_[4] = {(const float*)d_in[6], (const float*)d_in[7],
                           (const float*)d_in[8], (const float*)d_in[9]};
  int celld[4]; for (int i = 0; i < 4; ++i) celld[i] = in_sizes[6 + i] / Bn;

  const float* W1  = (const float*)d_in[10];
  const float* as1 = (const float*)d_in[11];
  const float* ad1 = (const float*)d_in[12];
  const float* b1  = (const float*)d_in[13];
  const float* W2  = (const float*)d_in[14];
  const float* as2 = (const float*)d_in[15];
  const float* ad2 = (const float*)d_in[16];
  const float* b2  = (const float*)d_in[17];
  const float* Wg  = (const float*)d_in[18];
  const float* bg  = (const float*)d_in[19];
  const float* fc1W = (const float*)d_in[44]; const float* fc1b = (const float*)d_in[45];
  const float* fc2W = (const float*)d_in[46]; const float* fc2b = (const float*)d_in[47];
  const float* fc3W = (const float*)d_in[48]; const float* fc3b = (const float*)d_in[49];
  const float* outW = (const float*)d_in[50]; const float* outb = (const float*)d_in[51];

  // ----- workspace carve (~245 MB total) -----
  size_t off = 0;
  auto carve = [&](size_t elems) -> float* {
    float* p = (float*)((char*)d_ws + off);
    off += ((elems * 4 + 255) / 256) * 256;
    return p;
  };
  float* H1   = carve((size_t)Nn * HD1);   // GAT1 linear out (reused as CN later)
  float* O1   = carve((size_t)Nn * HD1);   // GAT1 attention out
  float* Hb2  = carve((size_t)Nn * D);     // GAT2 linear out
  float* O2   = carve((size_t)Nn * D);     // GAT2 attention out
  float* ASR1 = carve((size_t)Nn * Hh1);
  float* ADT1 = carve((size_t)Nn * Hh1);
  float* M1   = carve((size_t)Nn * Hh1);
  float* S1   = carve((size_t)Nn * Hh1);
  float* ASR2 = carve(Nn); float* ADT2 = carve(Nn);
  float* M2   = carve(Nn); float* S2   = carve(Nn);
  float* POOL = carve((size_t)Bn * D);
  float* XC   = carve((size_t)Bn * 768);
  float* XCN  = carve((size_t)Bn * 768);
  float* TA   = carve((size_t)Bn * 2048);
  float* TB   = carve((size_t)Bn * 512);
  float* TC   = carve((size_t)Bn * D);
  float* CN   = H1;                        // cell phase runs after drug phase

  auto gemm = [&](const float* A, const float* W, const float* bias, float* C,
                  int M, int N, int K, int ldc, int colOff, int act) {
    dim3 g((N + BNT - 1) / BNT, (M + BMT - 1) / BMT);
    k_gemm<<<g, 256, 0, stream>>>(A, W, bias, C, M, N, K, ldc, colOff, act);
  };
  auto fill = [&](float* p, float v, int n) {
    k_fill<<<(n + 255) / 256, 256, 0, stream>>>(p, v, n);
  };

  // ================= drug branches =================
  for (int dg = 0; dg < 2; ++dg) {
    const float* x = x_[dg]; const int* ei = ei_[dg];
    // GAT layer 1 (78 -> 10x128)
    gemm(x, W1, nullptr, H1, Nn, HD1, F, HD1, 0, 0);
    k_att_scores<<<(Nn * Hh1 + 255) / 256, 256, 0, stream>>>(H1, as1, ad1, ASR1, ADT1, Nn, Hh1, D);
    fill(M1, -1e30f, Nn * Hh1);
    fill(S1, 0.f, Nn * Hh1);
    k_edge_max<<<(ET * Hh1 + 255) / 256, 256, 0, stream>>>(ei, E, ET, Hh1, ASR1, ADT1, M1);
    k_edge_sum<<<(ET * Hh1 + 255) / 256, 256, 0, stream>>>(ei, E, ET, Hh1, ASR1, ADT1, M1, S1);
    k_bias_rows<<<((size_t)Nn * HD1 + 255) / 256, 256, 0, stream>>>(O1, b1, Nn, HD1);
    k_edge_apply<<<ET * Hh1, 128, 0, stream>>>(ei, E, ET, Hh1, D, ASR1, ADT1, M1, S1, H1, O1);
    k_elu<<<((size_t)Nn * HD1 + 255) / 256, 256, 0, stream>>>(O1, Nn * HD1);
    // GAT layer 2 (1280 -> 128, 1 head)
    gemm(O1, W2, nullptr, Hb2, Nn, D, HD1, D, 0, 0);
    k_att_scores<<<(Nn + 255) / 256, 256, 0, stream>>>(Hb2, as2, ad2, ASR2, ADT2, Nn, 1, D);
    fill(M2, -1e30f, Nn);
    fill(S2, 0.f, Nn);
    k_edge_max<<<(ET + 255) / 256, 256, 0, stream>>>(ei, E, ET, 1, ASR2, ADT2, M2);
    k_edge_sum<<<(ET + 255) / 256, 256, 0, stream>>>(ei, E, ET, 1, ASR2, ADT2, M2, S2);
    k_bias_rows<<<((size_t)Nn * D + 255) / 256, 256, 0, stream>>>(O2, b2, Nn, D);
    k_edge_apply<<<ET, 128, 0, stream>>>(ei, E, ET, 1, D, ASR2, ADT2, M2, S2, Hb2, O2);
    k_elu<<<((size_t)Nn * D + 255) / 256, 256, 0, stream>>>(O2, Nn * D);
    // global max pool + relu(g @ Wg + bg) -> XC[:, dg*128 : dg*128+128]
    fill(POOL, -1e30f, Bn * D);
    k_pool_max<<<((size_t)Nn * D + 255) / 256, 256, 0, stream>>>(O2, bat_[dg], POOL, Nn, D);
    k_pool_fix<<<(Bn * D + 255) / 256, 256, 0, stream>>>(POOL, Bn * D);
    gemm(POOL, Wg, bg, XC, Bn, D, D, 768, dg * 128, 1);
  }

  // ================= cell expression branches =================
  for (int i = 0; i < 4; ++i) {
    const float* rW1 = (const float*)d_in[20 + i * 6 + 0];
    const float* rb1 = (const float*)d_in[20 + i * 6 + 1];
    const float* rW2 = (const float*)d_in[20 + i * 6 + 2];
    const float* rb2 = (const float*)d_in[20 + i * 6 + 3];
    const float* rW3 = (const float*)d_in[20 + i * 6 + 4];
    const float* rb3 = (const float*)d_in[20 + i * 6 + 5];
    k_l2norm<<<Bn, 256, 0, stream>>>(cell_[i], CN, celld[i]);
    gemm(CN, rW1, rb1, TA, Bn, 2048, celld[i], 2048, 0, 1);
    gemm(TA, rW2, rb2, TB, Bn, 512, 2048, 512, 0, 1);
    gemm(TB, rW3, rb3, XC, Bn, 128, 512, 768, 256 + i * 128, 1);
  }

  // ================= fused head =================
  k_l2norm<<<Bn, 256, 0, stream>>>(XC, XCN, 768);
  gemm(XCN, fc1W, fc1b, TA, Bn, 2048, 768, 2048, 0, 1);
  gemm(TA, fc2W, fc2b, TB, Bn, 512, 2048, 512, 0, 1);
  gemm(TB, fc3W, fc3b, TC, Bn, 128, 512, 128, 0, 1);
  k_out2<<<(Bn * 2 + 255) / 256, 256, 0, stream>>>(TC, outW, outb, (float*)d_out, Bn, 128, 2);
}
